// Qwen3MoeAttention_57329223467118
// MI455X (gfx1250) — matrix operand order repaired and checked by reading, not compiled, not run on hardware
//
#include <hip/hip_runtime.h>
#include <stdint.h>

// ---------------------------------------------------------------- constants
#define B_     2
#define S_     1024
#define HID_   2048
#define NH_    32
#define NKV_   4
#define D_     128
#define BLK_   32
#define MAXB_  128
#define NQ_    (NH_ * D_)     // 4096
#define NKVD_  (NKV_ * D_)    // 512
#define MROWS_ (B_ * S_)      // 2048
#define EPS_   1e-6f
#define SCALE_ 0.08838834764831845f  // 1/sqrt(128)

typedef __bf16 bf16_t;
typedef __attribute__((ext_vector_type(16))) __bf16    v16bf;
typedef __attribute__((ext_vector_type(8)))  float     v8f;
typedef __attribute__((ext_vector_type(4)))  unsigned  v4u;
typedef __attribute__((ext_vector_type(8)))  int       v8i;
typedef __attribute__((ext_vector_type(4)))  int       v4i;

union FragU { uint4 q[2]; v16bf v; };

// A-operand fragment (16x32 bf16, M x K): lane r holds row M=r,
// lane-half hi covers K = {hi*8..hi*8+7} U {16+hi*8..23+hi*8}.
__device__ inline v16bf frag_a(const bf16_t* p, int ldh, int r, int hi) {
  FragU u;
  u.q[0] = *reinterpret_cast<const uint4*>(p + r * ldh + hi * 8);
  u.q[1] = *reinterpret_cast<const uint4*>(p + r * ldh + 16 + hi * 8);
  return u.v;
}

// B-operand fragment (32x16 bf16, K x N) loaded from row-major [N][K]:
// lane r holds col N=r, lane-half hi covers K = {hi*8..hi*8+7} U {16+hi*8..23+hi*8}.
__device__ inline v16bf frag_b(const bf16_t* p, int ldh, int r, int hi) {
  FragU u;
  u.q[0] = *reinterpret_cast<const uint4*>(p + r * ldh + hi * 8);
  u.q[1] = *reinterpret_cast<const uint4*>(p + r * ldh + 16 + hi * 8);
  return u.v;
}

__device__ inline v8f v8f_zero() {
  v8f z;
#pragma unroll
  for (int i = 0; i < 8; ++i) z[i] = 0.f;
  return z;
}

__device__ inline v8f wmma_bf16(v16bf a, v16bf b, v8f c) {
  // (neg_a, A, neg_b, B, c_mod, C, reuse_a, reuse_b)
  return __builtin_amdgcn_wmma_f32_16x16x32_bf16(false, a, false, b,
                                                 (short)0, c, false, false);
}

// ---------------------------------------------------------------- TDM helper
// Issue a 2-D TENSOR_LOAD_TO_LDS of a [tile_d1 rows x tile_d0 cols] bf16 tile
// from a row-major [tensor_d1 x tensor_d0] tensor (row stride stride0, in
// elements). Tile is packed densely into LDS at lds_off.
// D# layout per CDNA5 ISA 8.3/8.4 (group0: count/lds/global/type,
// group1: data_size/dims/tile/strides; groups 2-3 zero for <=2-D).
// This toolchain exposes the 6-arg builtin:
//   (uint32x4 g0, int32x8 g1, int32x4 g2, int32x4 g3, int32x8 ext, i32 cpol)
__device__ inline void tdm_load_2d_bf16(uint32_t lds_off, const void* gptr,
                                        uint32_t tensor_d0, uint32_t tensor_d1,
                                        uint32_t tile_d0, uint32_t tile_d1,
                                        uint32_t stride0) {
  uint64_t ga = (uint64_t)(uintptr_t)gptr;
  v4u g0;
  g0[0] = 1u;                                          // count=1, no gather
  g0[1] = lds_off;                                     // lds_addr (bytes)
  g0[2] = (uint32_t)(ga & 0xFFFFFFFFu);                // global_addr[31:0]
  g0[3] = (uint32_t)((ga >> 32) & 0x01FFFFFFu)         // global_addr[56:32]
        | (2u << 30);                                  // type = 2 ("image")
  v8i g1;
  g1[0] = (int)(1u << 16);                             // data_size = 1 -> 2 bytes
  g1[1] = (int)((tensor_d0 & 0xFFFFu) << 16);          // tensor_dim0[15:0]
  g1[2] = (int)(((tensor_d0 >> 16) & 0xFFFFu)
        | ((tensor_d1 & 0xFFFFu) << 16));              // dim0 hi | dim1 lo
  g1[3] = (int)(((tensor_d1 >> 16) & 0xFFFFu)
        | ((tile_d0 & 0xFFFFu) << 16));                // dim1 hi | tile_dim0
  g1[4] = (int)(tile_d1 & 0xFFFFu);                    // tile_dim1 (tile_dim2=0)
  g1[5] = (int)stride0;                                // tensor_dim0_stride lo32
  g1[6] = 0;                                           // stride0 hi | dim1_stride lo
  g1[7] = 0;
  v4i g2  = {0, 0, 0, 0};
  v4i g3  = {0, 0, 0, 0};
  v8i ext = {0, 0, 0, 0, 0, 0, 0, 0};                  // unused for <=2-D
  __builtin_amdgcn_tensor_load_to_lds(g0, g1, g2, g3, ext, 0);
}

__device__ inline uint32_t lds_offset_of(const void* shared_ptr) {
  // generic pointers to LDS carry the LDS byte offset in their low 32 bits
  return (uint32_t)(uintptr_t)shared_ptr;
}

// ---------------------------------------------------------------- utilities
__global__ void cast_f32_bf16(const float* __restrict__ s,
                              bf16_t* __restrict__ d, int n) {
  int i = blockIdx.x * blockDim.x + threadIdx.x;
  int stride = gridDim.x * blockDim.x;
  for (; i < n; i += stride) d[i] = (bf16_t)s[i];
}

__global__ void zero_f32(float* __restrict__ p, int n) {
  int i = blockIdx.x * blockDim.x + threadIdx.x;
  int stride = gridDim.x * blockDim.x;
  for (; i < n; i += stride) p[i] = 0.f;
}

// ---------------------------------------------------------------- GEMM
// C[M,N] (f32) = A[M,K] (bf16, row-major) * W[N,K]^T (bf16, row-major).
// 256 threads = 8 wave32; 128x128 tile; K-step 32. Operand tiles staged by
// the Tensor Data Mover (dense [128][32] bf16 tiles in LDS).
__launch_bounds__(256)
__global__ void gemm_bt(const bf16_t* __restrict__ A,
                        const bf16_t* __restrict__ W,
                        float* __restrict__ C, int M, int N, int K) {
  __shared__ __align__(16) bf16_t As[128 * 32];
  __shared__ __align__(16) bf16_t Ws[128 * 32];
  const int tid  = threadIdx.x;
  const int lane = tid & 31;
  const int wave = tid >> 5;
  const int r    = lane & 15;
  const int hi   = lane >> 4;
  const int wm   = wave >> 2;   // 0..1  -> 64-row band
  const int wn   = wave & 3;    // 0..3  -> 32-col band
  const int m0   = blockIdx.y * 128;
  const int n0   = blockIdx.x * 128;

  v8f acc[4][2];
#pragma unroll
  for (int i = 0; i < 4; ++i)
#pragma unroll
    for (int j = 0; j < 2; ++j) acc[i][j] = v8f_zero();

  const uint32_t ldsA = lds_offset_of(&As[0]);
  const uint32_t ldsW = lds_offset_of(&Ws[0]);

  for (int k0 = 0; k0 < K; k0 += 32) {
    // ---- TDM staging: two 128x32 bf16 tiles, issued once per workgroup
    if (wave == 0) {
      tdm_load_2d_bf16(ldsA, A + (size_t)m0 * K + k0,
                       (uint32_t)K, (uint32_t)M, 32u, 128u, (uint32_t)K);
      tdm_load_2d_bf16(ldsW, W + (size_t)n0 * K + k0,
                       (uint32_t)K, (uint32_t)N, 32u, 128u, (uint32_t)K);
      __builtin_amdgcn_s_wait_tensorcnt(0);
    } else if (k0 + 32 < K) {
      // other waves prefetch the next K tiles into GL2 while the TDM runs
      const int prow = tid >> 1;
      const int pcol = (tid & 1) * 16;
      __builtin_prefetch(A + (size_t)(m0 + prow) * K + k0 + 32 + pcol, 0, 1);
      __builtin_prefetch(W + (size_t)(n0 + prow) * K + k0 + 32 + pcol, 0, 1);
    }
    __syncthreads();

    v16bf af[4], bf[2];
#pragma unroll
    for (int i = 0; i < 4; ++i) af[i] = frag_a(As + (wm * 64 + i * 16) * 32, 32, r, hi);
#pragma unroll
    for (int j = 0; j < 2; ++j) bf[j] = frag_b(Ws + (wn * 32 + j * 16) * 32, 32, r, hi);
#pragma unroll
    for (int i = 0; i < 4; ++i)
#pragma unroll
      for (int j = 0; j < 2; ++j) acc[i][j] = wmma_bf16(af[i], bf[j], acc[i][j]);
    __syncthreads();
  }

  // store: C element (e of v8f) -> row M = e + 8*hi, col N = r
#pragma unroll
  for (int i = 0; i < 4; ++i)
#pragma unroll
    for (int j = 0; j < 2; ++j)
#pragma unroll
      for (int e = 0; e < 8; ++e) {
        int row = m0 + wm * 64 + i * 16 + e + 8 * hi;
        int col = n0 + wn * 32 + j * 16 + r;
        C[(size_t)row * N + col] = acc[i][j][e];
      }
}

// ---------------------------------------------------------------- RMSNorm + RoPE (Q)
__launch_bounds__(128)
__global__ void norm_rope_q(const float* __restrict__ qf,
                            const float* __restrict__ w,
                            const float* __restrict__ cosT,
                            const float* __restrict__ sinT,
                            bf16_t* __restrict__ qb) {
  const int idx = blockIdx.x;          // (b*S + s)*NH + h
  const int h   = idx % NH_;
  const int row = idx / NH_;
  const int s   = row % S_;
  const int b   = row / S_;
  const int d   = threadIdx.x;

  __shared__ float red[128];
  __shared__ float ybuf[128];

  float x = qf[(size_t)row * NQ_ + h * D_ + d];
  red[d] = x * x;
  __syncthreads();
#pragma unroll
  for (int off = 64; off > 0; off >>= 1) {
    if (d < off) red[d] += red[d + off];
    __syncthreads();
  }
  float inv = rsqrtf(red[0] * (1.0f / D_) + EPS_);
  float y = x * inv * w[d];
  ybuf[d] = y;
  __syncthreads();
  float part = (d < 64) ? -ybuf[d + 64] : ybuf[d - 64];
  float o = y * cosT[s * D_ + d] + part * sinT[s * D_ + d];
  qb[((size_t)(b * NH_ + h) * S_ + s) * D_ + d] = (bf16_t)o;
}

// ---------------------------------------------------------------- RMSNorm + RoPE (K) + V + paged cache
__launch_bounds__(128)
__global__ void norm_rope_kv(const float* __restrict__ kf,
                             const float* __restrict__ vf,
                             const float* __restrict__ w,
                             const float* __restrict__ cosT,
                             const float* __restrict__ sinT,
                             const int*   __restrict__ pt,
                             bf16_t* __restrict__ kbb,
                             bf16_t* __restrict__ vbb,
                             float* __restrict__ cacheK,
                             float* __restrict__ cacheV) {
  const int idx = blockIdx.x;          // (b*S + s)*NKV + kv
  const int kv  = idx % NKV_;
  const int row = idx / NKV_;
  const int s   = row % S_;
  const int b   = row / S_;
  const int d   = threadIdx.x;

  __shared__ float red[128];
  __shared__ float ybuf[128];

  float x = kf[(size_t)row * NKVD_ + kv * D_ + d];
  red[d] = x * x;
  __syncthreads();
#pragma unroll
  for (int off = 64; off > 0; off >>= 1) {
    if (d < off) red[d] += red[d + off];
    __syncthreads();
  }
  float inv = rsqrtf(red[0] * (1.0f / D_) + EPS_);
  float y = x * inv * w[d];
  ybuf[d] = y;
  __syncthreads();
  float part = (d < 64) ? -ybuf[d + 64] : ybuf[d - 64];
  float kk = y * cosT[s * D_ + d] + part * sinT[s * D_ + d];
  float vv = vf[(size_t)row * NKVD_ + kv * D_ + d];

  const int blk = pt[b * (S_ / BLK_) + (s >> 5)];
  const size_t coff = (((size_t)blk * NKV_ + kv) * BLK_ + (s & 31)) * D_ + d;
  cacheK[coff] = kk;
  cacheV[coff] = vv;

  const size_t aoff = ((size_t)(b * NKV_ + kv) * S_ + s) * D_ + d;
  kbb[aoff] = (bf16_t)kk;
  vbb[aoff] = (bf16_t)vv;
}

// ---------------------------------------------------------------- causal flash attention (WMMA)
// grid: (S/128, NH, B). 256 threads = 8 waves; wave w owns 16 Q rows.
// K tile staged by the TDM; V tile staged manually (transposed).
__launch_bounds__(256)
__global__ void attn_kernel(const bf16_t* __restrict__ qb,
                            const bf16_t* __restrict__ kb,
                            const bf16_t* __restrict__ vb,
                            float* __restrict__ attnf) {
  const int qt  = blockIdx.x;
  const int h   = blockIdx.y;
  const int b   = blockIdx.z;
  const int kvh = h >> 3;              // GQA: 8 heads per KV head

  const int tid  = threadIdx.x;
  const int lane = tid & 31;
  const int wave = tid >> 5;
  const int r    = lane & 15;
  const int hi   = lane >> 4;
  const int qbase = qt * 128;
  const int qr0   = qbase + wave * 16;

  __shared__ __align__(16) bf16_t Ks[32 * 128];      // [key][d]
  __shared__ __align__(16) bf16_t Vts[128 * 32];     // [d][key] (transposed)
  __shared__ __align__(16) float  Sc[8][16][32];     // per-wave score tiles
  __shared__ __align__(16) bf16_t Pb[8][16][32];     // per-wave P tiles
  __shared__ float rowScale[8][16];
  __shared__ float rowL[8][16];

  const bf16_t* Q  = qb + (size_t)(b * NH_ + h)    * S_ * D_;
  const bf16_t* Kg = kb + (size_t)(b * NKV_ + kvh) * S_ * D_;
  const bf16_t* Vg = vb + (size_t)(b * NKV_ + kvh) * S_ * D_;

  // Q fragments (D=128 -> 4 K-steps of 32), loaded once from global
  v16bf aq[4];
#pragma unroll
  for (int ks = 0; ks < 4; ++ks)
    aq[ks] = frag_a(Q + (size_t)qr0 * D_ + ks * 32, D_, r, hi);

  v8f oacc[8];
#pragma unroll
  for (int i = 0; i < 8; ++i) oacc[i] = v8f_zero();

  float m_st = -3.0e38f, l_st = 0.0f;

  const int nblk   = (qbase + 128) >> 5;  // uniform over workgroup (causal bound)
  const int srow   = tid >> 3;            // staging: key row 0..31
  const int schunk = (tid & 7) * 16;      // staging: d-chunk (halves)
  const uint32_t ldsK = lds_offset_of(&Ks[0]);

  for (int j = 0; j < nblk; ++j) {
    const int kb0 = j * 32;
    // ---- TDM stages K tile (32 keys x 128 d); all waves transpose-stage V
    if (wave == 0)
      tdm_load_2d_bf16(ldsK, Kg + (size_t)kb0 * D_,
                       (uint32_t)D_, (uint32_t)S_, (uint32_t)D_, 32u,
                       (uint32_t)D_);
    {
      const bf16_t* vg = Vg + (size_t)(kb0 + srow) * D_ + schunk;
#pragma unroll
      for (int t = 0; t < 16; ++t) Vts[(schunk + t) * 32 + srow] = vg[t];
      if (j + 1 < nblk)
        __builtin_prefetch(Vg + (size_t)(kb0 + 32 + srow) * D_ + schunk, 0, 1);
    }
    if (wave == 0) __builtin_amdgcn_s_wait_tensorcnt(0);
    __syncthreads();

    // ---- scores: 16 (q rows) x 32 (keys) = 2 N-tiles x 4 K-steps of WMMA
    v8f sacc[2];
    sacc[0] = v8f_zero(); sacc[1] = v8f_zero();
#pragma unroll
    for (int nt = 0; nt < 2; ++nt)
#pragma unroll
      for (int ks = 0; ks < 4; ++ks) {
        v16bf bk = frag_b(Ks + (nt * 16) * 128 + ks * 32, 128, r, hi);
        sacc[nt] = wmma_bf16(aq[ks], bk, sacc[nt]);
      }

    // ---- scaled + causally masked scores -> LDS
#pragma unroll
    for (int nt = 0; nt < 2; ++nt)
#pragma unroll
      for (int e = 0; e < 8; ++e) {
        const int mrow = e + 8 * hi;
        const int key  = kb0 + nt * 16 + r;
        const int qrow = qr0 + mrow;
        float sv = sacc[nt][e] * SCALE_;
        if (key > qrow) sv = -3.0e38f;
        Sc[wave][mrow][nt * 16 + r] = sv;
      }
    __syncthreads();

    // ---- online softmax: lane pair (r, r+16) shares row r, 16 cols each
    float bm = -3.0e38f;
#pragma unroll
    for (int c = 0; c < 16; ++c) bm = fmaxf(bm, Sc[wave][r][hi * 16 + c]);
    bm = fmaxf(bm, __shfl_xor(bm, 16, 32));
    const float newm = fmaxf(m_st, bm);
    float psum = 0.f;
#pragma unroll
    for (int c = 0; c < 16; ++c) {
      float p = __expf(Sc[wave][r][hi * 16 + c] - newm);
      psum += p;
      Pb[wave][r][hi * 16 + c] = (bf16_t)p;
    }
    psum += __shfl_xor(psum, 16, 32);
    const float corr = __expf(m_st - newm);
    l_st = l_st * corr + psum;
    m_st = newm;
    if (hi == 0) rowScale[wave][r] = corr;
    __syncthreads();

    // ---- rescale O, accumulate P x V (8 d-tiles, K=32 keys)
    v16bf ap = frag_a(&Pb[wave][0][0], 32, r, hi);
#pragma unroll
    for (int nt = 0; nt < 8; ++nt) {
#pragma unroll
      for (int e = 0; e < 8; ++e) oacc[nt][e] *= rowScale[wave][e + 8 * hi];
      v16bf bv = frag_b(Vts + (nt * 16) * 32, 32, r, hi);
      oacc[nt] = wmma_bf16(ap, bv, oacc[nt]);
    }
    __syncthreads();   // before next iteration restages Ks/Vts and Sc
  }

  if (hi == 0) rowL[wave][r] = l_st;
  __syncthreads();

#pragma unroll
  for (int nt = 0; nt < 8; ++nt)
#pragma unroll
    for (int e = 0; e < 8; ++e) {
      const int mrow = e + 8 * hi;
      const float invl = 1.0f / rowL[wave][mrow];
      const int qrow = qr0 + mrow;
      const int dcol = nt * 16 + r;
      attnf[(size_t)(b * S_ + qrow) * NQ_ + h * D_ + dcol] = oacc[nt][e] * invl;
    }
}

// ---------------------------------------------------------------- launcher
extern "C" void kernel_launch(void* const* d_in, const int* in_sizes, int n_in,
                              void* d_out, int out_size, void* d_ws, size_t ws_size,
                              hipStream_t stream) {
  (void)in_sizes; (void)n_in; (void)out_size; (void)ws_size;

  const float* hidden = (const float*)d_in[0];
  const float* wq     = (const float*)d_in[1];
  const float* wk     = (const float*)d_in[2];
  const float* wv     = (const float*)d_in[3];
  const float* wo     = (const float*)d_in[4];
  const float* qnw    = (const float*)d_in[5];
  const float* knw    = (const float*)d_in[6];
  const float* cosT   = (const float*)d_in[7];
  const float* sinT   = (const float*)d_in[8];
  const int*   pt     = (const int*)d_in[9];

  float* out    = (float*)d_out;
  float* cacheK = out + (size_t)MROWS_ * HID_;
  float* cacheV = cacheK + (size_t)MAXB_ * NKV_ * BLK_ * D_;

  uint8_t* cur = (uint8_t*)d_ws;
  auto alloc = [&](size_t bytes) {
    void* p = cur;
    cur += (bytes + 255) & ~(size_t)255;
    return p;
  };

  bf16_t* hb    = (bf16_t*)alloc((size_t)MROWS_ * HID_ * 2);
  bf16_t* wqb   = (bf16_t*)alloc((size_t)NQ_ * HID_ * 2);
  bf16_t* wkb   = (bf16_t*)alloc((size_t)NKVD_ * HID_ * 2);
  bf16_t* wvb   = (bf16_t*)alloc((size_t)NKVD_ * HID_ * 2);
  bf16_t* wob   = (bf16_t*)alloc((size_t)HID_ * NQ_ * 2);
  float*  qf    = (float*) alloc((size_t)MROWS_ * NQ_ * 4);
  float*  kf    = (float*) alloc((size_t)MROWS_ * NKVD_ * 4);
  float*  vf    = (float*) alloc((size_t)MROWS_ * NKVD_ * 4);
  bf16_t* qb    = (bf16_t*)alloc((size_t)B_ * NH_ * S_ * D_ * 2);
  bf16_t* kbb   = (bf16_t*)alloc((size_t)B_ * NKV_ * S_ * D_ * 2);
  bf16_t* vbb   = (bf16_t*)alloc((size_t)B_ * NKV_ * S_ * D_ * 2);
  float*  attnf = (float*) alloc((size_t)MROWS_ * NQ_ * 4);
  bf16_t* attnb = (bf16_t*)alloc((size_t)MROWS_ * NQ_ * 2);

  // 1) downconvert activations + weights to bf16
  cast_f32_bf16<<<dim3(2048), 256, 0, stream>>>(hidden, hb, MROWS_ * HID_);
  cast_f32_bf16<<<dim3(4096), 256, 0, stream>>>(wq, wqb, NQ_ * HID_);
  cast_f32_bf16<<<dim3(1024), 256, 0, stream>>>(wk, wkb, NKVD_ * HID_);
  cast_f32_bf16<<<dim3(1024), 256, 0, stream>>>(wv, wvb, NKVD_ * HID_);
  cast_f32_bf16<<<dim3(4096), 256, 0, stream>>>(wo, wob, HID_ * NQ_);

  // 2) QKV projections (WMMA bf16 -> f32, TDM-staged tiles)
  gemm_bt<<<dim3(NQ_ / 128,   MROWS_ / 128), 256, 0, stream>>>(hb, wqb, qf, MROWS_, NQ_,   HID_);
  gemm_bt<<<dim3(NKVD_ / 128, MROWS_ / 128), 256, 0, stream>>>(hb, wkb, kf, MROWS_, NKVD_, HID_);
  gemm_bt<<<dim3(NKVD_ / 128, MROWS_ / 128), 256, 0, stream>>>(hb, wvb, vf, MROWS_, NKVD_, HID_);

  // 3) zero the paged cache outputs (unwritten blocks must be 0)
  zero_f32<<<dim3(2048), 256, 0, stream>>>(cacheK, 2 * MAXB_ * NKV_ * BLK_ * D_);

  // 4) RMSNorm + RoPE; cache scatter; bf16 repack for attention
  norm_rope_q <<<dim3(MROWS_ * NH_),  128, 0, stream>>>(qf, qnw, cosT, sinT, qb);
  norm_rope_kv<<<dim3(MROWS_ * NKV_), 128, 0, stream>>>(kf, vf, knw, cosT, sinT, pt,
                                                        kbb, vbb, cacheK, cacheV);

  // 5) causal flash attention (WMMA + TDM K staging)
  attn_kernel<<<dim3(S_ / 128, NH_, B_), 256, 0, stream>>>(qb, kbb, vbb, attnf);

  // 6) output projection
  cast_f32_bf16<<<dim3(4096), 256, 0, stream>>>(attnf, attnb, MROWS_ * NQ_);
  gemm_bt<<<dim3(HID_ / 128, MROWS_ / 128), 256, 0, stream>>>(attnb, wob, out, MROWS_, HID_, NQ_);
}
